// SwinTransformerLayer_31190052504437
// MI455X (gfx1250) — matrix operand order repaired and checked by reading, not compiled, not run on hardware
//
#include <hip/hip_runtime.h>
#include <hip/hip_bf16.h>

typedef __attribute__((ext_vector_type(16))) _Float16 v16h;
typedef __attribute__((ext_vector_type(8)))  float    v8f;

#define C_    192
#define HW_   65536     // 256*256
#define NWIN  1024      // 32*32 windows per batch image
#define XSTR  200       // f16 LDS row stride for [64 x 192] token-major buffers (+pad)
#define VSTR  80        // f16 LDS row stride for transposed V [192 x 64] (+pad, 16-half aligned)
#define SSTR  72        // LDS row stride for 64x64 score tiles (+pad)

__device__ __forceinline__ v8f wmma32(v16h a, v16h b, v8f c) {
  // D = A(16x32 f16) * B(32x16 f16) + C(16x16 f32)
  return __builtin_amdgcn_wmma_f32_16x16x32_f16(false, a, false, b, (short)0, c, false, false);
}

// A fragment 16x32 f16 from LDS, row-major buffer with stride `ld` (halfs).
// Layout: lanes 0-15 rows M=0..15 holding K {0..15}; lanes 16-31 same rows, K {16..31} (matches B loaders).
__device__ __forceinline__ v16h load_a_lds(const _Float16* buf, int ld, int row0, int k0, int lane) {
  int m  = row0 + (lane & 15);
  int kb = k0 + ((lane >> 4) << 4);
  const _Float16* p = buf + m * ld + kb;
  v16h a;
#pragma unroll
  for (int i = 0; i < 8; ++i) a[i] = p[i];
#pragma unroll
  for (int i = 0; i < 8; ++i) a[8 + i] = p[8 + i];
  return a;
}

// B fragment 32x16: B[k][n] = buf[(n0+n)*ld + k0+k]  (transposed row-major source, k contiguous)
__device__ __forceinline__ v16h load_b_rowT_lds(const _Float16* buf, int ld, int n0, int k0, int lane) {
  int n  = lane & 15;
  int kb = k0 + ((lane >> 4) << 4);
  const _Float16* p = buf + (n0 + n) * ld + kb;
  v16h b;
#pragma unroll
  for (int i = 0; i < 16; ++i) b[i] = p[i];
  return b;
}

// B fragment 32x16 from pre-converted f16 weight W[rows][ldk]: B[k][n] = W[n0+n][k0+k]
// 32-byte aligned -> single v16h (2x b128) load, no conversion VALU.
__device__ __forceinline__ v16h load_b_wth(const _Float16* __restrict__ w, int ldk,
                                           int n0, int k0, int lane) {
  int n  = lane & 15;
  int kb = k0 + ((lane >> 4) << 4);
  return *(const v16h*)(w + (n0 + n) * ldk + kb);
}

// boundary-mask region id along one axis (H=256, ws=8, shift=4): windows are 8-aligned,
// so only the last window row/col (index 31) splits into regions 1 (t<4) and 2 (t>=4).
__device__ __forceinline__ int regid(int wc, int tc) {
  return (wc < 31) ? 0 : ((tc < 4) ? 1 : 2);
}

// ---------------------------------------------------------------------------
// Kernel 0: one-shot f32 -> f16 weight conversion into workspace (L2 resident)
// ---------------------------------------------------------------------------
__global__ __launch_bounds__(256) void cvt_w_kernel(const float* __restrict__ qkv_w,
                                                    const float* __restrict__ proj_w,
                                                    _Float16* __restrict__ qkv_wh,
                                                    _Float16* __restrict__ proj_wh) {
  int i = blockIdx.x * 256 + threadIdx.x;
  if (i < 3 * C_ * C_) qkv_wh[i]  = (_Float16)qkv_w[i];
  if (i < C_ * C_)     proj_wh[i] = (_Float16)proj_w[i];
}

// ---------------------------------------------------------------------------
// Kernel 1/3: per-(b,c) instance-norm stats over 65536 spatial elements
// ---------------------------------------------------------------------------
__global__ __launch_bounds__(256) void in_stats_kernel(const float* __restrict__ src,
                                                       float* __restrict__ mean,
                                                       float* __restrict__ rstd) {
  __shared__ float s1[256], s2[256];
  const int pc = blockIdx.x;                 // b*C + c
  const float4* p = (const float4*)(src + ((long)pc << 16));
  float sum = 0.f, sum2 = 0.f;
  for (int i = threadIdx.x; i < HW_ / 4; i += 256) {
    float4 v = p[i];
    sum  += v.x + v.y + v.z + v.w;
    sum2 += v.x*v.x + v.y*v.y + v.z*v.z + v.w*v.w;
  }
  s1[threadIdx.x] = sum; s2[threadIdx.x] = sum2;
  __syncthreads();
  for (int o = 128; o > 0; o >>= 1) {
    if (threadIdx.x < o) { s1[threadIdx.x] += s1[threadIdx.x + o]; s2[threadIdx.x] += s2[threadIdx.x + o]; }
    __syncthreads();
  }
  if (threadIdx.x == 0) {
    float m   = s1[0] * (1.0f / 65536.0f);
    float var = s2[0] * (1.0f / 65536.0f) - m * m;
    mean[pc] = m;
    rstd[pc] = rsqrtf(var + 1e-5f);
  }
}

// ---------------------------------------------------------------------------
// Kernel 2: fused shifted-window attention, one workgroup (8 wave32) per window
// ---------------------------------------------------------------------------
__global__ __launch_bounds__(256) void swin_attn_kernel(
    const float* __restrict__ x,           const _Float16* __restrict__ qkv_wh,
    const float* __restrict__ qkv_b,       const _Float16* __restrict__ proj_wh,
    const float* __restrict__ proj_b,      const float* __restrict__ rel_bias,
    const float* __restrict__ mean1,       const float* __restrict__ rstd1,
    float* __restrict__ x1)
{
  __shared__ _Float16 Xs[64 * XSTR];   // normalized window, later reused as attention output O
  __shared__ _Float16 Qs[64 * XSTR];
  __shared__ _Float16 Ks[64 * XSTR];
  __shared__ _Float16 Vt[C_ * VSTR];   // V stored transposed: Vt[channel][token]
  __shared__ float    Ss[64 * SSTR];   // scores (f32)
  __shared__ _Float16 Ps[64 * SSTR];   // softmax probs (f16)

  const int tid  = threadIdx.x;
  const int lane = tid & 31;
  const int wave = tid >> 5;
  const int bi   = blockIdx.x;
  const int b    = bi >> 10;           // / 1024 windows per image
  const int w    = bi & 1023;
  const int wy   = w >> 5, wx = w & 31;

  // ---- Stage 1: gather shifted window, apply instance norm, store f16 ----
  for (int idx = tid; idx < 64 * C_; idx += 256) {
    int c  = idx >> 6;                 // / 64
    int t  = idx & 63;
    int ty = t >> 3, tx = t & 7;
    int gy = ((wy << 3) + ty + 4) & 255;   // roll(-shift): shifted[i] = x[i+shift mod H]
    int gx = ((wx << 3) + tx + 4) & 255;
    int pc = b * C_ + c;
    float v = x[((long)pc << 16) + (gy << 8) + gx];
    Xs[t * XSTR + c] = (_Float16)((v - mean1[pc]) * rstd1[pc]);
  }
  __syncthreads();

  // ---- Stage 2: QKV = Xs[64x192] @ qkv_w^T[192x576] + b  (144 WMMA tiles) ----
  const float qscale = 0.17677669529663687f;   // 1/sqrt(32)
  for (int tile = wave; tile < 144; tile += 8) {
    int mt = tile & 3;
    int n0 = (tile >> 2) << 4;
    v8f acc = {};
#pragma unroll
    for (int ks = 0; ks < 6; ++ks) {
      if (ks < 5) __builtin_prefetch(qkv_wh + (n0 + (lane & 15)) * C_ + ((ks + 1) << 5), 0, 1);
      v16h a  = load_a_lds(Xs, XSTR, mt << 4, ks << 5, lane);
      v16h bf = load_b_wth(qkv_wh, C_, n0, ks << 5, lane);
      acc = wmma32(a, bf, acc);
    }
    int col = n0 + (lane & 15);
    float bias = qkv_b[col];
    int rbase = (mt << 4) + ((lane >> 4) << 3);
    if (col < 2 * C_) {
#pragma unroll
      for (int v = 0; v < 8; ++v) {
        float val = acc[v] + bias;
        int r = rbase + v;
        if (col < C_) Qs[r * XSTR + col]        = (_Float16)(val * qscale);
        else          Ks[r * XSTR + (col - C_)] = (_Float16)val;
      }
    } else {
      // transposed V store: 8 consecutive halves per lane
      _Float16* pv = Vt + (col - 2 * C_) * VSTR + rbase;
#pragma unroll
      for (int v = 0; v < 8; ++v) pv[v] = (_Float16)(acc[v] + bias);
    }
  }
  __syncthreads();

  // ---- Stage 3: per-head attention (6 heads, d = 32) ----
  for (int h = 0; h < 6; ++h) {
    const int c0 = h << 5;

    // S = Q_h @ K_h^T : 16 tiles of 16x16, K-dim = 32 (1 WMMA each); 2 tiles/wave
#pragma unroll
    for (int j = 0; j < 2; ++j) {
      int st = wave * 2 + j;
      int mt = st >> 2, nt = st & 3;
      v16h a  = load_a_lds(Qs, XSTR, mt << 4, c0, lane);
      v16h bf = load_b_rowT_lds(Ks, XSTR, nt << 4, c0, lane);
      v8f acc = {};
      acc = wmma32(a, bf, acc);
      // fuse relative-position bias + shifted-window boundary mask at store
      int mcol = (nt << 4) + (lane & 15);
      int my = mcol >> 3, mx = mcol & 7;
      int rid_m = regid(wy, my) * 3 + regid(wx, mx);
      int rbase = (mt << 4) + ((lane >> 4) << 3);
#pragma unroll
      for (int v = 0; v < 8; ++v) {
        int r  = rbase + v;
        int ry = r >> 3, rx = r & 7;
        int rel = (ry - my + 7) * 15 + (rx - mx + 7);
        float bias  = rel_bias[rel * 6 + h];
        int   rid_r = regid(wy, ry) * 3 + regid(wx, rx);
        float msk   = (rid_r != rid_m) ? -100.0f : 0.0f;
        Ss[r * SSTR + mcol] = acc[v] + bias + msk;
      }
    }
    __syncthreads();

    // row softmax: 4 threads per row, quad reduction via lane shuffles
    {
      int r = tid >> 2;            // row 0..63
      int q = tid & 3;             // quarter of the row
      const float* row = Ss + r * SSTR + (q << 4);
      float e[16];
      float mx = row[0];
#pragma unroll
      for (int m = 1; m < 16; ++m) mx = fmaxf(mx, row[m]);
      mx = fmaxf(mx, __shfl_xor(mx, 1, 32));
      mx = fmaxf(mx, __shfl_xor(mx, 2, 32));
      float s = 0.f;
#pragma unroll
      for (int m = 0; m < 16; ++m) { e[m] = __expf(row[m] - mx); s += e[m]; }
      s += __shfl_xor(s, 1, 32);
      s += __shfl_xor(s, 2, 32);
      float inv = 1.0f / s;
      _Float16* prow = Ps + r * SSTR + (q << 4);
#pragma unroll
      for (int m = 0; m < 16; ++m) prow[m] = (_Float16)(e[m] * inv);
    }
    __syncthreads();

    // O_h = P[64x64] @ V_h[64x32] : 8 tiles, one per wave, 2 k-steps
    // B[k][n] = V[token k][c0+n] = Vt[c0+n][k]  -> contiguous LDS reads
    {
      int mt = wave >> 1, nt = wave & 1;
      v8f acc = {};
#pragma unroll
      for (int ks = 0; ks < 2; ++ks) {
        v16h a  = load_a_lds(Ps, SSTR, mt << 4, ks << 5, lane);
        v16h bf = load_b_rowT_lds(Vt, VSTR, c0 + (nt << 4), ks << 5, lane);
        acc = wmma32(a, bf, acc);
      }
      int col   = c0 + (nt << 4) + (lane & 15);
      int rbase = (mt << 4) + ((lane >> 4) << 3);
#pragma unroll
      for (int v = 0; v < 8; ++v)
        Xs[(rbase + v) * XSTR + col] = (_Float16)acc[v];   // Xs reused as O
    }
    __syncthreads();
  }

  // ---- Stage 4: proj + residual: x1 = x + (O @ proj_w^T + b), un-shift on store ----
  for (int tile = wave; tile < 48; tile += 8) {
    int mt = tile & 3;
    int n0 = (tile >> 2) << 4;
    v8f acc = {};
#pragma unroll
    for (int ks = 0; ks < 6; ++ks) {
      v16h a  = load_a_lds(Xs, XSTR, mt << 4, ks << 5, lane);
      v16h bf = load_b_wth(proj_wh, C_, n0, ks << 5, lane);
      acc = wmma32(a, bf, acc);
    }
    int col   = n0 + (lane & 15);
    float bias = proj_b[col];
    int rbase = (mt << 4) + ((lane >> 4) << 3);
#pragma unroll
    for (int v = 0; v < 8; ++v) {
      int t  = rbase + v;
      int ty = t >> 3, tx = t & 7;
      int gy = ((wy << 3) + ty + 4) & 255;   // roll(+shift) merge position
      int gx = ((wx << 3) + tx + 4) & 255;
      long off = ((long)(b * C_ + col) << 16) + (gy << 8) + gx;
      x1[off] = x[off] + acc[v] + bias;
    }
  }
}

// ---------------------------------------------------------------------------
// Kernel 4: FiLM branch: out = x1 + norm2(x1)*gamma + beta, gamma/beta from cond
// ---------------------------------------------------------------------------
__global__ __launch_bounds__(256) void film_kernel(
    const float* __restrict__ x1,    const float* __restrict__ cond,
    const float* __restrict__ film_w, const float* __restrict__ film_b,
    const float* __restrict__ mean2,  const float* __restrict__ rstd2,
    float* __restrict__ out)
{
  __shared__ float sg[128], sb[128];
  const int pc = blockIdx.x;
  const int b = pc / C_, c = pc - b * C_;
  const int t = threadIdx.x;
  if (t < 128) {
    float cv = cond[b * 128 + t];
    sg[t] = cv * film_w[c * 128 + t];
    sb[t] = cv * film_w[(C_ + c) * 128 + t];
  }
  __syncthreads();
  for (int o = 64; o > 0; o >>= 1) {
    if (t < o) { sg[t] += sg[t + o]; sb[t] += sb[t + o]; }
    __syncthreads();
  }
  const float gamma = sg[0] + film_b[c];
  const float beta  = sb[0] + film_b[C_ + c];
  const float m = mean2[pc], r = rstd2[pc];
  const float4* p = (const float4*)(x1 + ((long)pc << 16));
  float4*       q = (float4*)(out + ((long)pc << 16));
  const float gr = r * gamma;
  for (int i = t; i < HW_ / 4; i += 256) {
    float4 v = p[i], o4;
    o4.x = v.x + (v.x - m) * gr + beta;
    o4.y = v.y + (v.y - m) * gr + beta;
    o4.z = v.z + (v.z - m) * gr + beta;
    o4.w = v.w + (v.w - m) * gr + beta;
    q[i] = o4;
  }
}

extern "C" void kernel_launch(void* const* d_in, const int* in_sizes, int n_in,
                              void* d_out, int out_size, void* d_ws, size_t ws_size,
                              hipStream_t stream) {
  const float* x        = (const float*)d_in[0];
  const float* cond     = (const float*)d_in[1];
  const float* qkv_w    = (const float*)d_in[2];
  const float* qkv_b    = (const float*)d_in[3];
  const float* proj_w   = (const float*)d_in[4];
  const float* proj_b   = (const float*)d_in[5];
  const float* rel_bias = (const float*)d_in[6];
  const float* film_w   = (const float*)d_in[7];
  const float* film_b   = (const float*)d_in[8];
  float* out = (float*)d_out;

  const long BCHW = 4L * C_ * HW_;          // 50,331,648 floats (~201 MB)
  float* x1    = (float*)d_ws;
  float* mean1 = x1 + BCHW;
  float* rstd1 = mean1 + 4 * C_;
  float* mean2 = rstd1 + 4 * C_;
  float* rstd2 = mean2 + 4 * C_;
  _Float16* qkv_wh  = (_Float16*)(rstd2 + 4 * C_);
  _Float16* proj_wh = qkv_wh + 3 * C_ * C_;

  cvt_w_kernel<<<(3 * C_ * C_ + 255) / 256, 256, 0, stream>>>(qkv_w, proj_w, qkv_wh, proj_wh);
  in_stats_kernel<<<4 * C_, 256, 0, stream>>>(x, mean1, rstd1);
  swin_attn_kernel<<<4 * NWIN, 256, 0, stream>>>(x, qkv_wh, qkv_b, proj_wh, proj_b,
                                                 rel_bias, mean1, rstd1, x1);
  in_stats_kernel<<<4 * C_, 256, 0, stream>>>(x1, mean2, rstd2);
  film_kernel<<<4 * C_, 256, 0, stream>>>(x1, cond, film_w, film_b, mean2, rstd2, out);
}
